// GATModule_72670846648445
// MI455X (gfx1250) — compile-verified, matrix operand order checked by reading, not run on hardware
//
#include <hip/hip_runtime.h>
#include <hip/hip_bf16.h>

// ---------------------------------------------------------------------------
// GAT module, restructured for MI455X (gfx1250, wave32, WMMA):
//  * Dead-code elimination: only node_emb at the first node of each graph is
//    consumed, so only G*DEG = 32K of the 3.2M edges are computed (100x).
//  * One wave32 per graph: lane == edge (DEG == 32), segment softmax becomes
//    a pure intra-wave shuffle reduction.
//  * All MLP GEMMs run on v_wmma_f32_16x16x32_f16 (f16 in / f32 accumulate),
//    26 WMMAs per graph.
//  * Weights are converted to f16 AND pre-swizzled into WMMA B-fragment
//    lane order once (prep kernel), so each B fragment in the hot kernel is
//    a single contiguous 32-byte load per lane (2x global_load_b128) instead
//    of 16 strided 2-byte loads + half-packing movs. 13 KB total, L2-resident
//    for all 1000 waves.
// ---------------------------------------------------------------------------

typedef __attribute__((ext_vector_type(16))) _Float16 v16h;
typedef __attribute__((ext_vector_type(8)))  float    v8f;

// Pre-swizzled B fragments in d_ws, 512 halves (1 KB) each:
//  0..3 : W1a  [11->32 x 64], n0 = 16*f
//  4..5 : W1b  [64 x 16],     k0 = 32*(f-4)
//  6    : W2a  [16->32 x 16]
//  7..10: W3a  [16->32 x 64], n0 = 16*(f-7)
// 11..12: W3b  [64 x 16],     k0 = 32*(f-11)
#define FRAG_W1A 0
#define FRAG_W1B 4
#define FRAG_W2A 6
#define FRAG_W3A 7
#define FRAG_W3B 11
#define NUM_FRAGS 13

// K index held by element h (0..15) of a v16h fragment, per CDNA5 ISA 7.12.2:
// lanes 0-15:  VGPR0..3 -> K=0..7,  VGPR4..7 -> K=16..23
// lanes 16-31: VGPR0..3 -> K=8..15, VGPR4..7 -> K=24..31
__device__ __forceinline__ int wmma_kmap(int h, int hiHalf) {
  int v = h >> 1, p = h & 1;
  int k = (v < 4) ? (2 * v + p) : (16 + 2 * (v - 4) + p);
  return k + (hiHalf ? 8 : 0);
}

// Convert f32 weights to f16 and lay each 32x16 B tile out in exact fragment
// order: frag[lane*16 + h] = W[k0 + kmap(h, lane>=16)][n0 + lane%16].
__global__ void gat_prep_weights(const float* __restrict__ W1a,
                                 const float* __restrict__ W1b,
                                 const float* __restrict__ W2a,
                                 const float* __restrict__ W3a,
                                 const float* __restrict__ W3b,
                                 _Float16* __restrict__ ws) {
  int t = blockIdx.x * blockDim.x + threadIdx.x;
  int stride = gridDim.x * blockDim.x;
  for (int i = t; i < NUM_FRAGS * 512; i += stride) {
    int f    = i >> 9;          // fragment id
    int r    = i & 511;
    int lane = r >> 4;
    int h    = r & 15;
    int k    = wmma_kmap(h, lane >> 4);
    int n    = lane & 15;

    const float* src;
    int ldN, realK, k0 = 0, n0 = 0;
    if (f < 4)        { src = W1a; ldN = 64; realK = 11; n0 = 16 * f; }
    else if (f < 6)   { src = W1b; ldN = 16; realK = 64; k0 = 32 * (f - 4); }
    else if (f == 6)  { src = W2a; ldN = 16; realK = 16; }
    else if (f < 11)  { src = W3a; ldN = 64; realK = 16; n0 = 16 * (f - 7); }
    else              { src = W3b; ldN = 16; realK = 64; k0 = 32 * (f - 11); }

    int kk = k0 + k;
    ws[i] = (kk < realK) ? (_Float16)src[kk * ldN + n0 + n] : (_Float16)0.f;
  }
}

// A fragment: 16x32 tile of row-major f16 LDS matrix A (leading dim ld),
// rows m0..m0+15, cols k0..k0+31. K-contiguous halves -> ds_load_b128 pairs.
__device__ __forceinline__ v16h frag_a(const _Float16* A, int ld, int m0, int k0) {
  int lane = threadIdx.x & 31;
  int m = m0 + (lane & 15);
  int hi = lane >> 4;
  v16h a;
#pragma unroll
  for (int h = 0; h < 16; ++h) a[h] = A[m * ld + k0 + wmma_kmap(h, hi)];
  return a;
}

// B fragment: pre-swizzled, one contiguous 32-byte load per lane.
__device__ __forceinline__ v16h frag_b_pre(const _Float16* __restrict__ ws,
                                           int fragId) {
  int lane = threadIdx.x & 31;
  return *(const v16h*)(ws + fragId * 512 + lane * 16);
}

__device__ __forceinline__ v8f wmma_f16(v16h a, v16h b, v8f c) {
  return __builtin_amdgcn_wmma_f32_16x16x32_f16(false, a, false, b, (short)0, c,
                                                false, false);
}

__global__ void __launch_bounds__(32)
gat_graph_kernel(const float* __restrict__ x,
                 const int* __restrict__ edge_index,   // [2][E]
                 const float* __restrict__ edge_attr,  // [E][5]
                 const float* __restrict__ gammas,     // [G][2]
                 const _Float16* __restrict__ wsH,
                 const float* __restrict__ b1a, const float* __restrict__ b1b,
                 const float* __restrict__ b2a, const float* __restrict__ W2b,
                 const float* __restrict__ b2b,
                 const float* __restrict__ b3a, const float* __restrict__ b3b,
                 const float* __restrict__ W4a, const float* __restrict__ b4a,
                 const float* __restrict__ W4b, const float* __restrict__ b4b,
                 float* __restrict__ out, int M, int E) {
  __shared__ _Float16 Zh[32 * 32];    // z_ij, K padded to 32
  __shared__ _Float16 H1h[32 * 64];   // relu(Z@W1a+b1a)
  __shared__ _Float16 Qh[32 * 32];    // q_ij, K padded to 32
  __shared__ float    H2f[32 * 16];   // relu(Q@W2a+b2a)
  __shared__ _Float16 H3h[32 * 64];   // relu(Q@W3a+b3a)
  __shared__ float    MSGf[32 * 16];  // H3@W3b+b3b
  __shared__ float    attnS[32];
  __shared__ float    combS[18];

  const int g    = blockIdx.x;
  const int lane = threadIdx.x;          // one edge per lane (DEG == 32)
  const int node = g * M;                // first node of graph g
  const int e    = node * 32 + lane;     // its 32 contiguous out-edges

  // ---- build Z row: [x_src(3), x_dst(3), edge_attr(5), 0-pad] -------------
  {
    const int dst = edge_index[E + e];
    _Float16* zr = &Zh[lane * 32];
    zr[0] = (_Float16)x[node * 3 + 0];
    zr[1] = (_Float16)x[node * 3 + 1];
    zr[2] = (_Float16)x[node * 3 + 2];
    zr[3] = (_Float16)x[dst * 3 + 0];
    zr[4] = (_Float16)x[dst * 3 + 1];
    zr[5] = (_Float16)x[dst * 3 + 2];
#pragma unroll
    for (int k = 0; k < 5; ++k) zr[6 + k] = (_Float16)edge_attr[e * 5 + k];
#pragma unroll
    for (int k = 11; k < 32; ++k) zr[k] = (_Float16)0.f;
  }
  __syncthreads();

  const int hi = lane >> 4;
  const int nc = lane & 15;   // column this lane owns in C/D fragments

  // ---- L1: H1 = relu(Z @ W1a + b1a)  [32,64] ------------------------------
#pragma unroll
  for (int mt = 0; mt < 2; ++mt) {
    v16h a = frag_a(Zh, 32, mt * 16, 0);
#pragma unroll
    for (int nt = 0; nt < 4; ++nt) {
      v16h b = frag_b_pre(wsH, FRAG_W1A + nt);
      v8f c = {};
      c = wmma_f16(a, b, c);
      int n = nt * 16 + nc;
      float bias = b1a[n];
#pragma unroll
      for (int v = 0; v < 8; ++v) {
        int m = mt * 16 + hi * 8 + v;
        float val = c[v] + bias;
        H1h[m * 64 + n] = (_Float16)(val > 0.f ? val : 0.f);
      }
    }
  }
  __syncthreads();

  // ---- Q = H1 @ W1b + b1b  [32,16] (no relu) ------------------------------
#pragma unroll
  for (int mt = 0; mt < 2; ++mt) {
    v8f c = {};
#pragma unroll
    for (int kt = 0; kt < 2; ++kt) {
      v16h a = frag_a(H1h, 64, mt * 16, kt * 32);
      v16h b = frag_b_pre(wsH, FRAG_W1B + kt);
      c = wmma_f16(a, b, c);
    }
    float bias = b1b[nc];
#pragma unroll
    for (int v = 0; v < 8; ++v) {
      int m = mt * 16 + hi * 8 + v;
      Qh[m * 32 + nc] = (_Float16)(c[v] + bias);
    }
  }
#pragma unroll
  for (int k = 16; k < 32; ++k) Qh[lane * 32 + k] = (_Float16)0.f;  // K pad
  __syncthreads();

  // ---- H2 = relu(Q @ W2a + b2a)  [32,16] ----------------------------------
#pragma unroll
  for (int mt = 0; mt < 2; ++mt) {
    v16h a = frag_a(Qh, 32, mt * 16, 0);
    v16h b = frag_b_pre(wsH, FRAG_W2A);
    v8f c = {};
    c = wmma_f16(a, b, c);
    float bias = b2a[nc];
#pragma unroll
    for (int v = 0; v < 8; ++v) {
      int m = mt * 16 + hi * 8 + v;
      float val = c[v] + bias;
      H2f[m * 16 + nc] = val > 0.f ? val : 0.f;
    }
  }
  __syncthreads();

  // ---- raw logit + wave32 segment softmax (32 edges == 32 lanes) ----------
  {
    float raw = b2b[0];
#pragma unroll
    for (int k = 0; k < 16; ++k) raw += H2f[lane * 16 + k] * W2b[k];
    float mx = raw;
#pragma unroll
    for (int off = 16; off > 0; off >>= 1)
      mx = fmaxf(mx, __shfl_xor(mx, off, 32));
    float ex = __expf(raw - mx);
    float sm = ex;
#pragma unroll
    for (int off = 16; off > 0; off >>= 1) sm += __shfl_xor(sm, off, 32);
    attnS[lane] = ex / sm;
  }

  // ---- H3 = relu(Q @ W3a + b3a)  [32,64] ----------------------------------
#pragma unroll
  for (int mt = 0; mt < 2; ++mt) {
    v16h a = frag_a(Qh, 32, mt * 16, 0);
#pragma unroll
    for (int nt = 0; nt < 4; ++nt) {
      v16h b = frag_b_pre(wsH, FRAG_W3A + nt);
      v8f c = {};
      c = wmma_f16(a, b, c);
      int n = nt * 16 + nc;
      float bias = b3a[n];
#pragma unroll
      for (int v = 0; v < 8; ++v) {
        int m = mt * 16 + hi * 8 + v;
        float val = c[v] + bias;
        H3h[m * 64 + n] = (_Float16)(val > 0.f ? val : 0.f);
      }
    }
  }
  __syncthreads();

  // ---- MSG = H3 @ W3b + b3b  [32,16] --------------------------------------
#pragma unroll
  for (int mt = 0; mt < 2; ++mt) {
    v8f c = {};
#pragma unroll
    for (int kt = 0; kt < 2; ++kt) {
      v16h a = frag_a(H3h, 64, mt * 16, kt * 32);
      v16h b = frag_b_pre(wsH, FRAG_W3B + kt);
      c = wmma_f16(a, b, c);
    }
    float bias = b3b[nc];
#pragma unroll
    for (int v = 0; v < 8; ++v) {
      int m = mt * 16 + hi * 8 + v;
      MSGf[m * 16 + nc] = c[v] + bias;
    }
  }
  __syncthreads();

  // ---- node_emb = sum_e attn[e]*MSG[e]; combined = [emb(16), gamma(2)] ----
  if (lane < 16) {
    float acc = 0.f;
#pragma unroll
    for (int ee = 0; ee < 32; ++ee) acc += attnS[ee] * MSGf[ee * 16 + lane];
    combS[lane] = acc;
  } else if (lane < 18) {
    combS[lane] = gammas[g * 2 + (lane - 16)];
  }
  __syncthreads();

  // ---- MLP4: relu(combined @ W4a + b4a) @ W4b + b4b  -> out[g][2] ---------
  {
    float a0 = b4a[lane], a1 = b4a[lane + 32];
#pragma unroll
    for (int k = 0; k < 18; ++k) {
      float cv = combS[k];
      a0 += cv * W4a[k * 64 + lane];
      a1 += cv * W4a[k * 64 + lane + 32];
    }
    float h0 = a0 > 0.f ? a0 : 0.f;
    float h1 = a1 > 0.f ? a1 : 0.f;
    float p0 = h0 * W4b[lane * 2 + 0] + h1 * W4b[(lane + 32) * 2 + 0];
    float p1 = h0 * W4b[lane * 2 + 1] + h1 * W4b[(lane + 32) * 2 + 1];
#pragma unroll
    for (int off = 16; off > 0; off >>= 1) {
      p0 += __shfl_xor(p0, off, 32);
      p1 += __shfl_xor(p1, off, 32);
    }
    if (lane == 0) {
      out[g * 2 + 0] = p0 + b4b[0];
      out[g * 2 + 1] = p1 + b4b[1];
    }
  }
}

extern "C" void kernel_launch(void* const* d_in, const int* in_sizes, int n_in,
                              void* d_out, int out_size, void* d_ws, size_t ws_size,
                              hipStream_t stream) {
  const float* x         = (const float*)d_in[0];
  const int*   edge_index= (const int*)  d_in[1];
  const float* edge_attr = (const float*)d_in[2];
  // d_in[3] = batch: unused (graphs are contiguous, size M each)
  const float* gammas    = (const float*)d_in[4];
  const float* W1a = (const float*)d_in[5];  const float* b1a = (const float*)d_in[6];
  const float* W1b = (const float*)d_in[7];  const float* b1b = (const float*)d_in[8];
  const float* W2a = (const float*)d_in[9];  const float* b2a = (const float*)d_in[10];
  const float* W2b = (const float*)d_in[11]; const float* b2b = (const float*)d_in[12];
  const float* W3a = (const float*)d_in[13]; const float* b3a = (const float*)d_in[14];
  const float* W3b = (const float*)d_in[15]; const float* b3b = (const float*)d_in[16];
  const float* W4a = (const float*)d_in[17]; const float* b4a = (const float*)d_in[18];
  const float* W4b = (const float*)d_in[19]; const float* b4b = (const float*)d_in[20];

  const int N = in_sizes[0] / 3;
  const int E = in_sizes[2] / 5;
  const int G = in_sizes[4] / 2;
  const int M = N / G;

  _Float16* wsH = (_Float16*)d_ws;
  gat_prep_weights<<<8, 256, 0, stream>>>(W1a, W1b, W2a, W3a, W3b, wsH);
  gat_graph_kernel<<<G, 32, 0, stream>>>(x, edge_index, edge_attr, gammas, wsH,
                                         b1a, b1b, b2a, W2b, b2b, b3a, b3b,
                                         W4a, b4a, W4b, b4b,
                                         (float*)d_out, M, E);
}
